// MultiHeadAttention_51668456571275
// MI455X (gfx1250) — matrix operand order repaired and checked by reading, not compiled, not run on hardware
//
#include <hip/hip_runtime.h>
#include <hip/hip_bf16.h>

// ============================================================================
// MHA forward for MI455X (gfx1250, wave32, WMMA + TDM).
//  - All matrix math in V_WMMA_F32_16X16X32_F16 (f32 accumulate).
//  - Weights transposed+converted to f16 once -> GEMM B staging is a pure
//    contiguous 2D tile copy, done by the Tensor Data Mover when available.
//  - Flash attention: scores live in registers/LDS only.
// ============================================================================

typedef __attribute__((ext_vector_type(16))) _Float16 v16h;
typedef __attribute__((ext_vector_type(8)))  _Float16 v8h;
typedef __attribute__((ext_vector_type(8)))  float    v8f;
typedef __attribute__((ext_vector_type(4)))  unsigned v4u;
typedef __attribute__((ext_vector_type(8)))  int      v8i;
typedef __attribute__((ext_vector_type(4)))  int      v4i;

#define DMODEL 1024
#define NHEADS 16
#define DHEAD  64
#define SEQLEN 2048

#if defined(__AMDGCN__) && __has_builtin(__builtin_amdgcn_tensor_load_to_lds) && \
    __has_builtin(__builtin_amdgcn_s_wait_tensorcnt)
#define HAVE_TDM 1
#else
#define HAVE_TDM 0
#endif

static __device__ __forceinline__ v16h cat8(v8h lo, v8h hi) {
  return __builtin_shufflevector(lo, hi, 0,1,2,3,4,5,6,7,8,9,10,11,12,13,14,15);
}

// ---------------------------------------------------------------------------
// Kernel 1: transpose + convert weights: in f32 [K][N] -> out f16 [N][K]
// ---------------------------------------------------------------------------
__global__ void transpose_cvt_kernel(const float* __restrict__ in,
                                     _Float16* __restrict__ out, int N, int K) {
  __shared__ float t[32][33];
  const int tx = threadIdx.x, ty = threadIdx.y;
  const int x = blockIdx.x * 32 + tx;            // n (read col)
  #pragma unroll
  for (int j = 0; j < 4; ++j) {
    const int y = blockIdx.y * 32 + ty + j * 8;  // k (read row)
    t[ty + j * 8][tx] = in[(size_t)y * N + x];
  }
  __syncthreads();
  const int xo = blockIdx.y * 32 + tx;           // k (write col)
  #pragma unroll
  for (int j = 0; j < 4; ++j) {
    const int yo = blockIdx.x * 32 + ty + j * 8; // n (write row)
    out[(size_t)yo * K + xo] = (_Float16)t[tx][ty + j * 8];
  }
}

// ---------------------------------------------------------------------------
// Kernel 2: tiled WMMA GEMM  C[M,N] = A[M,K] @ B[K,N] + bias
//   B is given PRE-TRANSPOSED f16: Bt[N][K].
//   A_F16   : 0 -> A is f32 (converted during staging), 1 -> A is f16
//   OUT_MODE: 0 -> f16 [M,N] ; 1 -> f16 per-head transposed [B,H,Dh,L] ; 2 -> f32
// Workgroup: 256 threads (8 waves); tile 128x128; wave tile 32x64; K-step 32.
// Dynamic LDS layout: As f16[128][32] @ 0 (8KB), Bts f16[128][32] @ 8192 (8KB).
// ---------------------------------------------------------------------------
#define LDS_BTS_OFF 8192u

template <int A_F16, int OUT_MODE>
__global__ void wmma_gemm_kernel(const void* __restrict__ Ap,
                                 const _Float16* __restrict__ Bt,
                                 const float* __restrict__ bias,
                                 void* __restrict__ Cp,
                                 int M, int N, int K, int L) {
  extern __shared__ __align__(128) char smem[];
  _Float16 (*As)[32]  = reinterpret_cast<_Float16(*)[32]>(smem);
  _Float16 (*Bts)[32] = reinterpret_cast<_Float16(*)[32]>(smem + LDS_BTS_OFF);

  const int tid  = threadIdx.x;
  const int lane = tid & 31;
  const int w    = tid >> 5;         // 0..7
  const int rw   = w >> 1;           // 0..3 -> rows 32*rw
  const int cw   = w & 1;            // 0..1 -> cols 64*cw
  const int m0   = blockIdx.y * 128;
  const int n0   = blockIdx.x * 128;

  const float*    Af32 = (const float*)Ap;
  const _Float16* Af16 = (const _Float16*)Ap;

  v8f acc[2][4] = {};

  const int sr = tid >> 1;           // 0..127 (staging row)
  const int cb = (tid & 1) * 16;     // 0 or 16 (staging k-chunk)

  for (int k0 = 0; k0 < K; k0 += 32) {
    // ---- stage A tile 128x32 f16 (convert from f32 if needed) ----
    {
      v16h hv;
      if (A_F16) {
        hv = *(const v16h*)(Af16 + (size_t)(m0 + sr) * K + k0 + cb);
      } else {
        const float* g = Af32 + (size_t)(m0 + sr) * K + k0 + cb;
        #pragma unroll
        for (int j = 0; j < 16; ++j) hv[j] = (_Float16)g[j];
      }
      *(v16h*)&As[sr][cb] = hv;
    }

    // ---- stage B tile: Bt[n0..n0+127][k0..k0+31] -> LDS (pure 2D copy) ----
#if HAVE_TDM
    if (tid < 32) {  // wave 0 drives the Tensor Data Mover
      const unsigned long long ga =
          (unsigned long long)(const void*)(Bt + (size_t)n0 * K + k0);
      v4u g0;
      g0[0] = 1u;                                   // count=1, user descriptor
      g0[1] = LDS_BTS_OFF;                          // lds_addr (bytes)
      g0[2] = (unsigned)ga;                         // global_addr[31:0]
      g0[3] = (unsigned)((ga >> 32) & 0x1FFFFFFull) // global_addr[56:32]
              | (2u << 30);                         // type=2 ("image")
      v8i g1;
      g1[0] = (1 << 16);                            // data_size=2B; no mask
      g1[1] = (int)((K & 0xFFFF) << 16);            // tensor_dim0[15:0]
      g1[2] = (int)(((unsigned)K >> 16) | ((N & 0xFFFF) << 16)); // d0 hi, d1 lo
      g1[3] = (int)(((unsigned)N >> 16) | (32u << 16));          // d1 hi, tile_dim0=32
      g1[4] = 128;                                  // tile_dim1=128, tile_dim2=0
      g1[5] = K;                                    // tensor_dim0_stride[31:0]
      g1[6] = 0;                                    // stride hi, dim1_stride lo
      g1[7] = 0;
      v4i g2 = {0, 0, 0, 0}, g3 = {0, 0, 0, 0};     // unused (2D tensor)
      v8i g4 = {0, 0, 0, 0, 0, 0, 0, 0};            // unused trailing group
      __builtin_amdgcn_tensor_load_to_lds(g0, g1, g2, g3, g4, 0);
      __builtin_amdgcn_s_wait_tensorcnt(0);
    }
#else
    {
      v16h bv = *(const v16h*)(Bt + (size_t)(n0 + sr) * K + k0 + cb);
      *(v16h*)&Bts[sr][cb] = bv;
    }
#endif
    __syncthreads();

    // ---- WMMA: wave computes 32x64 (2 m-tiles x 4 n-tiles) ----
    const int aoff = (lane < 16) ? 0 : 16;
    const int boff = (lane < 16) ? 0 : 16;
    v16h a[2];
    #pragma unroll
    for (int mi = 0; mi < 2; ++mi) {
      const int arow = (rw << 5) + (mi << 4) + (lane & 15);
      a[mi] = *(const v16h*)&As[arow][aoff];
    }
    #pragma unroll
    for (int nt = 0; nt < 4; ++nt) {
      const int brow = (cw << 6) + (nt << 4) + (lane & 15);
      v16h b = *(const v16h*)&Bts[brow][boff];
      #pragma unroll
      for (int mi = 0; mi < 2; ++mi) {
        acc[mi][nt] = __builtin_amdgcn_wmma_f32_16x16x32_f16(
            false, a[mi], false, b, (short)0, acc[mi][nt], false, false);
      }
    }
    __syncthreads();
  }

  // ---- epilogue ----
  const int lhalf = lane >> 4;
  #pragma unroll
  for (int nt = 0; nt < 4; ++nt) {
    const int gn = n0 + (cw << 6) + (nt << 4) + (lane & 15);
    const float bv = bias[gn];
    #pragma unroll
    for (int mi = 0; mi < 2; ++mi) {
      #pragma unroll
      for (int v = 0; v < 8; ++v) {
        const int gm = m0 + (rw << 5) + (mi << 4) + v + (lhalf << 3);
        const float val = acc[mi][nt][v] + bv;
        if (OUT_MODE == 0) {
          ((_Float16*)Cp)[(size_t)gm * N + gn] = (_Float16)val;
        } else if (OUT_MODE == 1) {
          const int b   = gm / L;
          const int tok = gm - b * L;
          const int h   = gn >> 6;
          const int d   = gn & 63;
          ((_Float16*)Cp)[(((size_t)b * NHEADS + h) * DHEAD + d) * L + tok] =
              (_Float16)val;
        } else {
          ((float*)Cp)[(size_t)gm * N + gn] = val;
        }
      }
    }
  }
}

// ---------------------------------------------------------------------------
// Kernel 3: causal flash attention. One wave per 16-query tile.
//  Qh,Kh : f16 [B, L, 1024] (head-contiguous rows)
//  VhT   : f16 [B, H, Dh, L]
//  Oo    : f16 [B, L, 1024]
// ---------------------------------------------------------------------------
__global__ void flash_attn_kernel(const _Float16* __restrict__ Qh,
                                  const _Float16* __restrict__ Kh,
                                  const _Float16* __restrict__ VhT,
                                  _Float16* __restrict__ Oo, int L) {
  __shared__ __attribute__((aligned(64))) _Float16 Pl[16][32];

  const int lane  = threadIdx.x;
  const int lhalf = lane >> 4;
  const int qt = blockIdx.x;
  const int h  = blockIdx.y;
  const int b  = blockIdx.z;
  const int qi0 = qt << 4;

  const _Float16* Qb = Qh + (size_t)b * L * DMODEL + h * DHEAD;
  const _Float16* Kb = Kh + (size_t)b * L * DMODEL + h * DHEAD;
  const _Float16* Vb = VhT + ((size_t)b * NHEADS + h) * DHEAD * L;
  _Float16*       Ob = Oo + (size_t)b * L * DMODEL + h * DHEAD;

  // Q A-fragments (16x32 f16, two K-chunks over Dh=64)
  const int qrow = qi0 + (lane & 15);
  v16h aQ[2];
  #pragma unroll
  for (int c = 0; c < 2; ++c) {
    const int off = c * 32 + ((lane < 16) ? 0 : 16);
    aQ[c] = *(const v16h*)(Qb + (size_t)qrow * DMODEL + off);
  }

  float mrow[8], lrow[8];
  #pragma unroll
  for (int v = 0; v < 8; ++v) { mrow[v] = -3.0e38f; lrow[v] = 0.0f; }
  v8f acc[4] = {};

  const int jmax = (qi0 + 15) >> 5;
  const int kcol = lane & 15;
  const int kofB = (lane < 16) ? 0 : 16;

  for (int j = 0; j <= jmax; ++j) {
    const int key0 = j << 5;
    if (j < jmax)
      __builtin_prefetch(Kb + (size_t)(key0 + 32 + kcol) * DMODEL, 0, 1);

    // S = Q @ K^T (two 16x16 tiles)
    v8f S0 = {}, S1 = {};
    #pragma unroll
    for (int c = 0; c < 2; ++c) {
      const int doff = c * 32 + kofB;
      v16h bK0 = *(const v16h*)(Kb + (size_t)(key0 + kcol) * DMODEL + doff);
      v16h bK1 = *(const v16h*)(Kb + (size_t)(key0 + 16 + kcol) * DMODEL + doff);
      S0 = __builtin_amdgcn_wmma_f32_16x16x32_f16(false, aQ[c], false, bK0,
                                                  (short)0, S0, false, false);
      S1 = __builtin_amdgcn_wmma_f32_16x16x32_f16(false, aQ[c], false, bK1,
                                                  (short)0, S1, false, false);
    }
    S0 = S0 * 0.125f;
    S1 = S1 * 0.125f;

    if (j == jmax) {  // only the diagonal block can violate causality
      #pragma unroll
      for (int v = 0; v < 8; ++v) {
        const int qv = qi0 + v + (lhalf << 3);
        if (key0 + kcol      > qv) S0[v] = -3.0e38f;
        if (key0 + kcol + 16 > qv) S1[v] = -3.0e38f;
      }
    }

    // online softmax (row reductions across 16-lane halves)
    #pragma unroll
    for (int v = 0; v < 8; ++v) {
      float s0 = S0[v], s1 = S1[v];
      float bm = fmaxf(s0, s1);
      bm = fmaxf(bm, __shfl_xor(bm, 1, 32));
      bm = fmaxf(bm, __shfl_xor(bm, 2, 32));
      bm = fmaxf(bm, __shfl_xor(bm, 4, 32));
      bm = fmaxf(bm, __shfl_xor(bm, 8, 32));
      const float mnew = fmaxf(mrow[v], bm);
      const float p0 = __expf(s0 - mnew);
      const float p1 = __expf(s1 - mnew);
      float ps = p0 + p1;
      ps += __shfl_xor(ps, 1, 32);
      ps += __shfl_xor(ps, 2, 32);
      ps += __shfl_xor(ps, 4, 32);
      ps += __shfl_xor(ps, 8, 32);
      const float alpha = __expf(mrow[v] - mnew);
      lrow[v] = lrow[v] * alpha + ps;
      mrow[v] = mnew;
      #pragma unroll
      for (int n = 0; n < 4; ++n) acc[n][v] *= alpha;
      const int r = v + (lhalf << 3);
      Pl[r][kcol]      = (_Float16)p0;
      Pl[r][kcol + 16] = (_Float16)p1;
    }

    // relayout P: C-layout (LDS) -> A-fragment (16x32, K=keys)
    asm volatile("s_wait_dscnt 0" ::: "memory");
    {
      const int prow = lane & 15;
      const int poff = (lane < 16) ? 0 : 16;
      v16h aP = *(const v16h*)&Pl[prow][poff];

      // O += P @ V (V pre-transposed: contiguous along keys)
      #pragma unroll
      for (int n = 0; n < 4; ++n) {
        const int d = (n << 4) + kcol;
        v16h bV = *(const v16h*)(Vb + (size_t)d * L + key0 + kofB);
        acc[n] = __builtin_amdgcn_wmma_f32_16x16x32_f16(
            false, aP, false, bV, (short)0, acc[n], false, false);
      }
    }
    asm volatile("s_wait_dscnt 0" ::: "memory");
  }

  // normalize + store
  #pragma unroll
  for (int v = 0; v < 8; ++v) {
    const float inv = 1.0f / lrow[v];
    const int q = qi0 + v + (lhalf << 3);
    #pragma unroll
    for (int n = 0; n < 4; ++n) {
      Ob[(size_t)q * DMODEL + (n << 4) + kcol] = (_Float16)(acc[n][v] * inv);
    }
  }
}

// ---------------------------------------------------------------------------
// Host-side orchestration
// ---------------------------------------------------------------------------
extern "C" void kernel_launch(void* const* d_in, const int* in_sizes, int n_in,
                              void* d_out, int out_size, void* d_ws, size_t ws_size,
                              hipStream_t stream) {
  const float* q  = (const float*)d_in[0];
  const float* k  = (const float*)d_in[1];
  const float* v  = (const float*)d_in[2];
  const float* Wq = (const float*)d_in[3];
  const float* bq = (const float*)d_in[4];
  const float* Wk = (const float*)d_in[5];
  const float* bk = (const float*)d_in[6];
  const float* Wv = (const float*)d_in[7];
  const float* bv = (const float*)d_in[8];
  const float* Wo = (const float*)d_in[9];
  const float* bo = (const float*)d_in[10];

  const int M = in_sizes[0] / DMODEL;   // B * L = 8192
  const int L = SEQLEN;
  const int Bn = M / L;

  // workspace (f16): 4 transposed weights (8MB) + Qh/Kh/VhT/attnO (64MB)
  char* ws = (char*)d_ws;
  const size_t WSZ = (size_t)DMODEL * DMODEL * sizeof(_Float16);
  const size_t XSZ = (size_t)M * DMODEL * sizeof(_Float16);
  _Float16* WqT = (_Float16*)(ws);
  _Float16* WkT = (_Float16*)(ws + WSZ);
  _Float16* WvT = (_Float16*)(ws + 2 * WSZ);
  _Float16* WoT = (_Float16*)(ws + 3 * WSZ);
  _Float16* Qh  = (_Float16*)(ws + 4 * WSZ);
  _Float16* Kh  = (_Float16*)(ws + 4 * WSZ + XSZ);
  _Float16* VhT = (_Float16*)(ws + 4 * WSZ + 2 * XSZ);
  _Float16* AtO = (_Float16*)(ws + 4 * WSZ + 3 * XSZ);

  // 1) weights -> f16, transposed to [N][K]
  dim3 tgrid(DMODEL / 32, DMODEL / 32), tblk(32, 8);
  transpose_cvt_kernel<<<tgrid, tblk, 0, stream>>>(Wq, WqT, DMODEL, DMODEL);
  transpose_cvt_kernel<<<tgrid, tblk, 0, stream>>>(Wk, WkT, DMODEL, DMODEL);
  transpose_cvt_kernel<<<tgrid, tblk, 0, stream>>>(Wv, WvT, DMODEL, DMODEL);
  transpose_cvt_kernel<<<tgrid, tblk, 0, stream>>>(Wo, WoT, DMODEL, DMODEL);

  // 2) projections (WMMA GEMMs, 128x128 tiles, 16KB dynamic LDS)
  dim3 ggrid(DMODEL / 128, M / 128);
  const size_t gshm = 16384;
  wmma_gemm_kernel<0, 0><<<ggrid, 256, gshm, stream>>>(q, WqT, bq, Qh, M, DMODEL, DMODEL, L);
  wmma_gemm_kernel<0, 0><<<ggrid, 256, gshm, stream>>>(k, WkT, bk, Kh, M, DMODEL, DMODEL, L);
  wmma_gemm_kernel<0, 1><<<ggrid, 256, gshm, stream>>>(v, WvT, bv, VhT, M, DMODEL, DMODEL, L);

  // 3) causal flash attention
  dim3 fgrid(L / 16, NHEADS, Bn);
  flash_attn_kernel<<<fgrid, 32, 0, stream>>>(Qh, Kh, VhT, AtO, L);

  // 4) output projection -> f32 d_out (+bias)
  wmma_gemm_kernel<1, 2><<<ggrid, 256, gshm, stream>>>(AtO, WoT, bo, (float*)d_out,
                                                       M, DMODEL, DMODEL, L);
}
